// RadarEncoderBlock_55018531062256
// MI455X (gfx1250) — matrix operand order repaired and checked by reading, not compiled, not run on hardware
//
#include <hip/hip_runtime.h>
#include <hip/hip_bf16.h>

// ---------------------------------------------------------------------------
// TrajGRU (HKO-7) forward for gfx1250. Every conv is an implicit GEMM tiled
// for v_wmma_f32_16x16x32_bf16 (bf16 inputs, f32 accumulation).
//
// Block = 128 threads (4 waves, wave32). Block tile: 64 (M) x TN (N), TN=32|64.
// Per 32-wide K chunk: stage A (im2col gather, f32->bf16 packed pairs) and
// B^T (weights, [n][k]) in LDS as u32 words; each wave builds one A fragment
// and TN/16 B fragments (8 aligned ds_load_b32 each) and issues TN/16 WMMAs.
// Boundary handling is branch-free: clamped address + select, so EXEC stays
// full (WMMA requires EXEC all-ones) and loads pipeline without exec churn.
// ---------------------------------------------------------------------------

typedef __attribute__((ext_vector_type(16))) __bf16 v16bf;
typedef __attribute__((ext_vector_type(8)))  float  v8f;

#define MODE_DOWN 0
#define MODE_I2H  1
#define MODE_I2F  2
#define MODE_H2F  3
#define MODE_FLOW 4
#define MODE_RET  5

__device__ __forceinline__ unsigned bf16bits(float f) {
  unsigned u = __builtin_bit_cast(unsigned, f);
  u += 0x7FFFu + ((u >> 16) & 1u);          // round-to-nearest-even
  return u >> 16;
}
__device__ __forceinline__ unsigned pack_bf16x2(float lo, float hi) {
  return bf16bits(lo) | (bf16bits(hi) << 16);
}
__device__ __forceinline__ int clampi(int v, int lim) {  // [0, lim-1]
  v = v < 0 ? 0 : v;
  return v > lim - 1 ? lim - 1 : v;
}

// Branch-free im2col gather: A(m, k). m = b*4096 + oy*64 + ox.
template<int MODE>
__device__ __forceinline__ float gatherA(const float* __restrict__ src,
                                         int m, int k, int t) {
  const int b  = m >> 12;
  const int p  = m & 4095;
  const int oy = p >> 6;
  const int ox = p & 63;
  if constexpr (MODE == MODE_DOWN) {
    // x: (8, 12, 128, 128); k = c*16 + ky*4 + kx; stride 2, pad 1
    int c = k >> 4, r = k & 15;
    int iy = 2 * oy - 1 + (r >> 2);
    int ix = 2 * ox - 1 + (r & 3);
    bool ok = ((unsigned)iy < 128u) && ((unsigned)ix < 128u);
    float v = src[((b * 12 + c) << 14) + (clampi(iy, 128) << 7) + clampi(ix, 128)];
    return ok ? v : 0.0f;
  } else if constexpr (MODE == MODE_RET) {
    // warped (8, 320, 64, 64); 1x1 conv
    return src[((b * 320 + k) << 12) + p];
  } else {
    // 3x3 pad1 (I2H) or 5x5 pad2 (I2F/H2F/FLOW) over 64x64 images.
    int KS  = (MODE == MODE_I2H) ? 3 : 5;
    int PAD = (MODE == MODE_I2H) ? 1 : 2;
    int KK  = KS * KS;
    int c = k / KK, r = k - KK * c;
    int ry = r / KS;
    int iy = oy - PAD + ry;
    int ix = ox - PAD + (r - KS * ry);
    bool ok = ((unsigned)iy < 64u) && ((unsigned)ix < 64u);
    int chan;
    if constexpr (MODE == MODE_I2H || MODE == MODE_I2F) chan = b * 768 + t * 64 + c;
    else if constexpr (MODE == MODE_H2F)                chan = b * 64 + c;
    else                                                chan = b * 32 + c;   // FLOW
    float v = src[(chan << 12) + (clampi(iy, 64) << 6) + clampi(ix, 64)];
    return ok ? v : 0.0f;
  }
}

// K: reduction length (multiple of 32). OC: real #output channels.
// TN: N tile (32 or 64).
template<int MODE, int K, int OC, int TN>
__global__ __launch_bounds__(128) void conv_gemm(
    const float* __restrict__ src, const float* __restrict__ Wf,
    const float* __restrict__ bias, float* __restrict__ out, int t) {
  constexpr int NB = TN / 16;                 // accumulators per wave
  __shared__ unsigned sA[64][20];             // 64 rows x 16 k-words (+pad)
  __shared__ unsigned sBt[TN][20];            // TN rows (n) x 16 k-words (+pad)

  const int tid  = threadIdx.x;
  const int lane = tid & 31;
  const int wave = tid >> 5;
  const int q    = lane >> 4;                 // lane half
  const int ln   = lane & 15;
  const int row0 = blockIdx.x * 64;
  const int n0   = blockIdx.y * TN;

  v8f acc[NB];
#pragma unroll
  for (int nb = 0; nb < NB; ++nb) acc[nb] = (v8f){};

  union Frag { v16bf v; unsigned u[8]; };

  for (int kk = 0; kk < K; kk += 32) {
    __syncthreads();
    // Stage A tile: 1024 u32 words (2048 bf16), 8 words per thread.
#pragma unroll
    for (int i = 0; i < 8; ++i) {
      int e = tid + i * 128;
      int r = e >> 4, wi = e & 15;
      int m = row0 + r, kg = kk + wi * 2;
      float v0 = gatherA<MODE>(src, m, kg, t);
      float v1 = gatherA<MODE>(src, m, kg + 1, t);
      sA[r][wi] = pack_bf16x2(v0, v1);
    }
    // Stage B^T tile: TN*16 u32 words; weights are OIHW => row-major [OC][K].
#pragma unroll
    for (int i = 0; i < TN / 8; ++i) {
      int e = tid + i * 128;
      int n = e >> 4, wi = e & 15;
      int ng = n0 + n, kg = kk + wi * 2;
      float w0 = (ng < OC) ? Wf[(size_t)ng * K + kg] : 0.0f;
      float w1 = (ng < OC) ? Wf[(size_t)ng * K + kg + 1] : 0.0f;
      sBt[n][wi] = pack_bf16x2(w0, w1);
    }
    __syncthreads();

    // A fragment (ISA 16-bit 16x32 layout): VGPR v holds k-pairs at
    // (v<4?0:16) + q*8 + (v&3)*2  ->  word (v<4?0:8) + q*4 + (v&3).
    Frag a;
    const unsigned* arow = sA[wave * 16 + ln];
#pragma unroll
    for (int v = 0; v < 8; ++v)
      a.u[v] = arow[((v < 4) ? 0 : 8) + q * 4 + (v & 3)];

    // B fragments: same K layout as A, N = ln. With B^T in LDS,
    // VGPR v is word (v<4?0:8) + q*4 + (v&3) of row n.
#pragma unroll
    for (int nb = 0; nb < NB; ++nb) {
      Frag bfr;
      const unsigned* brow = sBt[nb * 16 + ln];
#pragma unroll
      for (int v = 0; v < 8; ++v)
        bfr.u[v] = brow[((v < 4) ? 0 : 8) + q * 4 + (v & 3)];
      acc[nb] = __builtin_amdgcn_wmma_f32_16x16x32_bf16(
          false, a.v, false, bfr.v, (short)0, acc[nb], false, false);
    }
  }

  // Epilogue. C/D layout: VGPR r -> M = r + 8*q, N = lane%16.
#pragma unroll
  for (int nb = 0; nb < NB; ++nb) {
    int n = n0 + nb * 16 + ln;
    if (n < OC) {
      float bv = bias[n];
#pragma unroll
      for (int r = 0; r < 8; ++r) {
        int m = row0 + wave * 16 + r + q * 8;
        int b = m >> 12, p = m & 4095;
        float v = acc[nb][r] + bv;
        float* dst = out + (((size_t)(b * OC + n)) << 12) + p;
        if constexpr (MODE == MODE_DOWN) {
          *dst = (v >= 0.0f) ? v : 0.2f * v;                 // LeakyReLU
        } else if constexpr (MODE == MODE_H2F) {
          float s = *dst + v;                                // += i2f, then leaky
          *dst = (s >= 0.0f) ? s : 0.2f * s;
        } else {
          *dst = v;
        }
      }
    }
  }
}

// Bilinear multi-trajectory warp of h_prev by -flows (zero padding).
__global__ void warp_kernel(const float* __restrict__ hprev,
                            const float* __restrict__ flows,
                            float* __restrict__ warped) {
  int idx = blockIdx.x * blockDim.x + threadIdx.x;   // B*L*F*4096
  if (idx >= 8 * 5 * 64 * 4096) return;
  int p = idx & 4095;
  int c = idx >> 12;          // b*320 + l*64 + f
  int f = c & 63;
  int bl = c >> 6;            // b*5 + l
  int l = bl % 5;
  int b = bl / 5;
  int y = p >> 6, x = p & 63;
  float fx = -flows[((b * 10 + 2 * l) << 12) + p];
  float fy = -flows[((b * 10 + 2 * l + 1) << 12) + p];
  float X = (float)x + fx;
  float Y = (float)y + fy;
  float x0f = floorf(X), y0f = floorf(Y);
  int x0 = (int)x0f, y0 = (int)y0f;
  float wx = X - x0f, wy = Y - y0f;
  const float* img = hprev + (((size_t)(b * 64 + f)) << 12);
  auto samp = [&](int xi, int yi) -> float {
    bool ok = ((unsigned)xi < 64u) && ((unsigned)yi < 64u);
    float v = img[(clampi(yi, 64) << 6) + clampi(xi, 64)];
    return ok ? v : 0.0f;
  };
  float va = samp(x0, y0), vb = samp(x0, y0 + 1);
  float vc = samp(x0 + 1, y0), vd = samp(x0 + 1, y0 + 1);
  warped[idx] = (1.0f - wx) * (1.0f - wy) * va + (1.0f - wx) * wy * vb +
                wx * (1.0f - wy) * vc + wx * wy * vd;
}

// GRU gating: r/u sigmoid, leaky candidate, blend; writes h_new and the
// t-slice of the (B,T,F,h,w) output.
__global__ void gate_kernel(const float* __restrict__ i2h,
                            const float* __restrict__ h2h,
                            const float* __restrict__ hprev,
                            float* __restrict__ hnew,
                            float* __restrict__ out, int t) {
  int idx = blockIdx.x * blockDim.x + threadIdx.x;   // B*64*4096
  if (idx >= 8 * 64 * 4096) return;
  int p = idx & 4095;
  int c = idx >> 12;          // b*64 + f
  int f = c & 63;
  int b = c >> 6;
  size_t base = ((size_t)(b * 192)) << 12;
  float ir = i2h[base + ((size_t)f << 12) + p];
  float iu = i2h[base + ((size_t)(64 + f) << 12) + p];
  float im = i2h[base + ((size_t)(128 + f) << 12) + p];
  float hr = h2h[base + ((size_t)f << 12) + p];
  float hu = h2h[base + ((size_t)(64 + f) << 12) + p];
  float hm = h2h[base + ((size_t)(128 + f) << 12) + p];
  float r = 1.0f / (1.0f + expf(-(ir + hr)));
  float u = 1.0f / (1.0f + expf(-(iu + hu)));
  float m = im + r * hm;
  m = (m >= 0.0f) ? m : 0.2f * m;
  float hp = hprev[idx];
  float hn = u * hp + (1.0f - u) * m;
  hnew[idx] = hn;
  out[(((size_t)((b * 12 + t) * 64 + f)) << 12) + p] = hn;
}

__global__ void zero_kernel(float* __restrict__ p, int n) {
  int i = blockIdx.x * blockDim.x + threadIdx.x;
  if (i < n) p[i] = 0.0f;
}

extern "C" void kernel_launch(void* const* d_in, const int* in_sizes, int n_in,
                              void* d_out, int out_size, void* d_ws, size_t ws_size,
                              hipStream_t stream) {
  const float* x      = (const float*)d_in[0];
  const float* W_down = (const float*)d_in[1];
  const float* b_down = (const float*)d_in[2];
  const float* W_i2h  = (const float*)d_in[3];
  const float* b_i2h  = (const float*)d_in[4];
  const float* W_i2f  = (const float*)d_in[5];
  const float* b_i2f  = (const float*)d_in[6];
  const float* W_h2f  = (const float*)d_in[7];
  const float* b_h2f  = (const float*)d_in[8];
  const float* W_flow = (const float*)d_in[9];
  const float* b_flow = (const float*)d_in[10];
  const float* W_ret  = (const float*)d_in[11];
  const float* b_ret  = (const float*)d_in[12];
  float* out = (float*)d_out;

  char* ws = (char*)d_ws;
  size_t off = 0;
  auto take = [&](size_t bytes) -> float* {
    char* p = ws + off;
    off = (off + bytes + 255) & ~(size_t)255;
    return (float*)p;
  };
  float* xd     = take((size_t)8 * 768 * 4096 * 4);  // downsampled stack
  float* i2h_t  = take((size_t)8 * 192 * 4096 * 4);  // per-step i2h
  float* f1     = take((size_t)8 * 32  * 4096 * 4);  // i2f + h2f, leaky
  float* flows  = take((size_t)8 * 10  * 4096 * 4);  // L flow fields
  float* warped = take((size_t)8 * 320 * 4096 * 4);  // L warped copies
  float* h2h    = take((size_t)8 * 192 * 4096 * 4);  // ret projection
  float* hA     = take((size_t)8 * 64  * 4096 * 4);  // hidden ping
  float* hB     = take((size_t)8 * 64  * 4096 * 4);  // hidden pong

  const dim3 blk(128);
  zero_kernel<<<(8 * 64 * 4096) / 256, 256, 0, stream>>>(hA, 8 * 64 * 4096);

  // Downsample: M=32768, K=192, N=768, fused LeakyReLU.
  conv_gemm<MODE_DOWN, 192, 768, 64><<<dim3(512, 12), blk, 0, stream>>>(
      x, W_down, b_down, xd, 0);

  float* hp = hA;
  float* hn = hB;
  for (int t = 0; t < 12; ++t) {
    conv_gemm<MODE_I2H, 576, 192, 64><<<dim3(512, 3), blk, 0, stream>>>(
        xd, W_i2h, b_i2h, i2h_t, t);
    conv_gemm<MODE_I2F, 1600, 32, 32><<<dim3(512, 1), blk, 0, stream>>>(
        xd, W_i2f, b_i2f, f1, t);
    conv_gemm<MODE_H2F, 1600, 32, 32><<<dim3(512, 1), blk, 0, stream>>>(
        hp, W_h2f, b_h2f, f1, t);               // f1 = leaky(i2f + h2f)
    conv_gemm<MODE_FLOW, 800, 10, 32><<<dim3(512, 1), blk, 0, stream>>>(
        f1, W_flow, b_flow, flows, t);
    warp_kernel<<<(8 * 5 * 64 * 4096) / 256, 256, 0, stream>>>(hp, flows, warped);
    conv_gemm<MODE_RET, 320, 192, 64><<<dim3(512, 3), blk, 0, stream>>>(
        warped, W_ret, b_ret, h2h, t);
    gate_kernel<<<(8 * 64 * 4096) / 256, 256, 0, stream>>>(
        i2h_t, h2h, hp, hn, out, t);
    float* tmp = hp; hp = hn; hn = tmp;
  }

  (void)in_sizes; (void)n_in; (void)out_size; (void)ws_size;
}
